// MoELayer_37383395344888
// MI455X (gfx1250) — compile-verified, matrix operand order checked by reading, not run on hardware
//
#include <hip/hip_runtime.h>

#define DIM 1024
#define HID 512
#define N_EXPERTS 16
#define TOP_K 4
#define T_TOKENS 2048
#define MAX_ROWS_PER_EXPERT 2048

typedef __attribute__((ext_vector_type(16))) __bf16 v16bf;
typedef __attribute__((ext_vector_type(8)))  float  v8f;

__device__ __forceinline__ v8f vzero8() { v8f v = {}; return v; }

__device__ __forceinline__ v8f wmma_bf16(v16bf a, v16bf b, v8f c) {
  // D = A(16x32 bf16) * B(32x16 bf16) + C(16x16 f32)
  return __builtin_amdgcn_wmma_f32_16x16x32_bf16(
      /*neg_a=*/false, a, /*neg_b=*/false, b,
      /*c_mod=*/(short)0, c, /*reuse_a=*/false, /*reuse_b=*/false);
}

// Low 32 bits of a generic shared-aperture address == LDS byte offset
// (ISA 10.2).
__device__ __forceinline__ unsigned lds_offset(const void* p) {
  return (unsigned)(unsigned long long)p;
}

// CDNA5 async direct global->LDS copy (no VGPR round trip), ASYNCcnt-tracked.
__device__ __forceinline__ void async_load_b32_to_lds(unsigned ldsOff, const void* g) {
  asm volatile("global_load_async_to_lds_b32 %0, %1, off"
               :: "v"(ldsOff), "v"(g)
               : "memory");
}

__device__ __forceinline__ void wait_asynccnt0() {
  asm volatile("s_wait_asynccnt 0x0" ::: "memory");
}

// Build a 16x32 (MxK) bf16 fragment for one lane from a row-major matrix.
// Per ISA 7.12.2: lanes 0-15 hold rows M=0..15 with K = {kh*8+2i} U {16+kh*8+2i};
// contiguous within each half, so these fold into b128 loads.
__device__ __forceinline__ v16bf load_frag(const __bf16* __restrict__ base,
                                           int stride, int row, int k0, int kh) {
  v16bf v;
  const __bf16* p = base + (size_t)row * stride + k0;
#pragma unroll
  for (int i = 0; i < 8; ++i) {
    const int k = (i < 4) ? (kh * 8 + 2 * i) : (16 + kh * 8 + 2 * (i - 4));
    v[2 * i]     = p[k];
    v[2 * i + 1] = p[k + 1];
  }
  return v;
}

__global__ void init_counts(int* __restrict__ cnt) {
  if (threadIdx.x < N_EXPERTS) cnt[threadIdx.x] = 0;
}

__global__ void cvt_f32_bf16(const float* __restrict__ s, __bf16* __restrict__ d, int n) {
  int i = blockIdx.x * blockDim.x + threadIdx.x;
  const int stride = gridDim.x * blockDim.x;
  for (; i < n; i += stride) d[i] = (__bf16)s[i];
}

// One block per token: 16 logits -> softmax -> top-4 -> renormalize -> gather lists.
__global__ void __launch_bounds__(256) router_kernel(
    const float* __restrict__ x, const float* __restrict__ gate_w,
    int* __restrict__ cnt, int* __restrict__ rowTok, float* __restrict__ rowW) {
  __shared__ float partial[N_EXPERTS][16];
  __shared__ float probs[N_EXPERTS];
  const int t = blockIdx.x;
  const float* xr = x + (size_t)t * DIM;
  const int e = threadIdx.x >> 4;
  const int s = threadIdx.x & 15;
  float sum = 0.f;
  for (int k = s; k < DIM; k += 16) sum += xr[k] * gate_w[e * DIM + k];
  partial[e][s] = sum;
  __syncthreads();
  if (threadIdx.x < N_EXPERTS) {
    float v = 0.f;
#pragma unroll
    for (int i = 0; i < 16; ++i) v += partial[threadIdx.x][i];
    probs[threadIdx.x] = v;
  }
  __syncthreads();
  if (threadIdx.x == 0) {
    float p[N_EXPERTS];
    float mx = -1e30f;
#pragma unroll
    for (int i = 0; i < N_EXPERTS; ++i) mx = fmaxf(mx, probs[i]);
    float den = 0.f;
#pragma unroll
    for (int i = 0; i < N_EXPERTS; ++i) { p[i] = __expf(probs[i] - mx); den += p[i]; }
    const float inv = 1.f / den;
#pragma unroll
    for (int i = 0; i < N_EXPERTS; ++i) p[i] *= inv;
    int idx[TOP_K]; float w[TOP_K]; float wsum = 0.f;
#pragma unroll
    for (int k = 0; k < TOP_K; ++k) {
      int bj = 0; float bv = -1.f;
#pragma unroll
      for (int i = 0; i < N_EXPERTS; ++i) {
        if (p[i] > bv) { bv = p[i]; bj = i; }
      }
      idx[k] = bj; w[k] = bv; wsum += bv; p[bj] = -2.f;
    }
    const float winv = 1.f / wsum;
    for (int k = 0; k < TOP_K; ++k) {
      const int pos = atomicAdd(&cnt[idx[k]], 1);
      rowTok[idx[k] * MAX_ROWS_PER_EXPERT + pos] = t;
      rowW [idx[k] * MAX_ROWS_PER_EXPERT + pos] = w[k] * winv;
    }
  }
}

// Shared FFN: one block per 16 tokens; out = (silu(x@sw1^T) * (x@sw3^T)) @ sw2^T
// 256 threads = 8 waves; GEMM1: wave covers 64 hid cols (4 tiles);
// GEMM2: wave covers 128 out cols (8 tiles). Plain stores (first writer of out).
__global__ void __launch_bounds__(256) shared_ffn_kernel(
    const __bf16* __restrict__ xb, const __bf16* __restrict__ w1b,
    const __bf16* __restrict__ w3b, const __bf16* __restrict__ w2b,
    float* __restrict__ out) {
  __shared__ __align__(16) __bf16 actLds[16 * HID];
  __shared__ __align__(16) __bf16 aTile[16 * 32];
  const int t0 = blockIdx.x * 16;
  const int tid = threadIdx.x;
  const int wave = tid >> 5;
  const int lane = tid & 31;
  const int kh = lane >> 4;
  const int r = lane & 15;

  // Per-thread staging assignment: 2 bf16 (4 bytes) of the 16x32 A tile.
  const int srr = (tid * 2) >> 5;
  const int scc = (tid * 2) & 31;
  const unsigned stageOff = lds_offset(&aTile[srr * 32 + scc]);

  v8f acc1[4], acc3[4];
#pragma unroll
  for (int j = 0; j < 4; ++j) { acc1[j] = vzero8(); acc3[j] = vzero8(); }

  for (int k0 = 0; k0 < DIM; k0 += 32) {
    __syncthreads();
    // Async direct-to-LDS stage of x tile rows (ASYNCcnt path).
    async_load_b32_to_lds(stageOff, xb + (size_t)(t0 + srr) * DIM + k0 + scc);
    __builtin_prefetch(xb + (size_t)(t0 + srr) * DIM + k0 + 32 + scc, 0, 3);
    wait_asynccnt0();
    __syncthreads();
    const v16bf a = load_frag(aTile, 32, r, 0, kh);
#pragma unroll
    for (int j = 0; j < 4; ++j) {
      const int nb = wave * 64 + j * 16;
      const v16bf b1 = load_frag(w1b, DIM, nb + r, k0, kh);
      const v16bf b3 = load_frag(w3b, DIM, nb + r, k0, kh);
      acc1[j] = wmma_bf16(a, b1, acc1[j]);
      acc3[j] = wmma_bf16(a, b3, acc3[j]);
    }
  }
  __syncthreads();
#pragma unroll
  for (int j = 0; j < 4; ++j) {
    const int col = wave * 64 + j * 16 + r;
#pragma unroll
    for (int g = 0; g < 8; ++g) {
      const int m = g + kh * 8;
      const float h1 = acc1[j][g];
      const float h3 = acc3[j][g];
      const float av = (h1 / (1.f + __expf(-h1))) * h3;  // silu(h1)*h3
      actLds[m * HID + col] = (__bf16)av;
    }
  }
  __syncthreads();

  v8f acc[8];
#pragma unroll
  for (int j = 0; j < 8; ++j) acc[j] = vzero8();
  for (int k0 = 0; k0 < HID; k0 += 32) {
    const v16bf a = load_frag(actLds, HID, r, k0, kh);
#pragma unroll
    for (int j = 0; j < 8; ++j) {
      const int nb = wave * 128 + j * 16;
      const v16bf b = load_frag(w2b, HID, nb + r, k0, kh);
      acc[j] = wmma_bf16(a, b, acc[j]);
    }
  }
#pragma unroll
  for (int j = 0; j < 8; ++j) {
    const int col = wave * 128 + j * 16 + r;
#pragma unroll
    for (int g = 0; g < 8; ++g) {
      const int m = g + kh * 8;
      out[(size_t)(t0 + m) * DIM + col] = acc[j][g];
    }
  }
}

// Routed experts: grid (rowTile, expert), early-exit past this expert's count.
// Gathers 16 tokens; fused GEMM1 -> silu*mul -> GEMM2; weighted scatter-add.
__global__ void __launch_bounds__(256) moe_expert_kernel(
    const __bf16* __restrict__ xb, const __bf16* __restrict__ w1b,
    const __bf16* __restrict__ w3b, const __bf16* __restrict__ w2b,
    const int* __restrict__ cnt, const int* __restrict__ rowTok,
    const float* __restrict__ rowW, float* __restrict__ out) {
  const int e = blockIdx.y;
  const int n = cnt[e];
  const int rt = blockIdx.x;
  if (rt * 16 >= n) return;

  __shared__ __align__(16) __bf16 actLds[16 * HID];
  __shared__ __align__(16) __bf16 aTile[16 * 32];
  __shared__ int   tokLds[16];
  __shared__ float wtLds[16];

  const int tid = threadIdx.x;
  const int wave = tid >> 5;
  const int lane = tid & 31;
  const int kh = lane >> 4;
  const int r = lane & 15;

  if (tid < 16) {
    const int row = rt * 16 + tid;
    if (row < n) {
      tokLds[tid] = rowTok[e * MAX_ROWS_PER_EXPERT + row];
      wtLds[tid]  = rowW [e * MAX_ROWS_PER_EXPERT + row];
    } else {
      tokLds[tid] = rowTok[e * MAX_ROWS_PER_EXPERT];  // valid pad token
      wtLds[tid]  = 0.f;                              // contributes nothing
    }
  }
  __syncthreads();

  const int srr = (tid * 2) >> 5;
  const int scc = (tid * 2) & 31;
  const unsigned stageOff = lds_offset(&aTile[srr * 32 + scc]);
  const __bf16* srcRow = xb + (size_t)tokLds[srr] * DIM + scc;  // gathered token row

  const __bf16* W1 = w1b + (size_t)e * HID * DIM;
  const __bf16* W3 = w3b + (size_t)e * HID * DIM;
  const __bf16* W2 = w2b + (size_t)e * DIM * HID;

  v8f acc1[4], acc3[4];
#pragma unroll
  for (int j = 0; j < 4; ++j) { acc1[j] = vzero8(); acc3[j] = vzero8(); }

  for (int k0 = 0; k0 < DIM; k0 += 32) {
    __syncthreads();
    // Async direct-to-LDS stage of gathered token rows (ASYNCcnt path).
    async_load_b32_to_lds(stageOff, srcRow + k0);
    __builtin_prefetch(srcRow + k0 + 32, 0, 3);
    wait_asynccnt0();
    __syncthreads();
    const v16bf a = load_frag(aTile, 32, r, 0, kh);
#pragma unroll
    for (int j = 0; j < 4; ++j) {
      const int nb = wave * 64 + j * 16;
      const v16bf b1 = load_frag(W1, DIM, nb + r, k0, kh);
      const v16bf b3 = load_frag(W3, DIM, nb + r, k0, kh);
      acc1[j] = wmma_bf16(a, b1, acc1[j]);
      acc3[j] = wmma_bf16(a, b3, acc3[j]);
    }
  }
  __syncthreads();
#pragma unroll
  for (int j = 0; j < 4; ++j) {
    const int col = wave * 64 + j * 16 + r;
#pragma unroll
    for (int g = 0; g < 8; ++g) {
      const int m = g + kh * 8;
      const float h1 = acc1[j][g];
      const float h3 = acc3[j][g];
      const float av = (h1 / (1.f + __expf(-h1))) * h3;
      actLds[m * HID + col] = (__bf16)av;
    }
  }
  __syncthreads();

  v8f acc[8];
#pragma unroll
  for (int j = 0; j < 8; ++j) acc[j] = vzero8();
  for (int k0 = 0; k0 < HID; k0 += 32) {
    const v16bf a = load_frag(actLds, HID, r, k0, kh);
#pragma unroll
    for (int j = 0; j < 8; ++j) {
      const int nb = wave * 128 + j * 16;
      const v16bf b = load_frag(W2, HID, nb + r, k0, kh);
      acc[j] = wmma_bf16(a, b, acc[j]);
    }
  }
#pragma unroll
  for (int j = 0; j < 8; ++j) {
    const int col = wave * 128 + j * 16 + r;
#pragma unroll
    for (int g = 0; g < 8; ++g) {
      const int m = g + kh * 8;
      const float val = acc[j][g] * wtLds[m];
      atomicAdd(&out[(size_t)tokLds[m] * DIM + col], val);
    }
  }
}

extern "C" void kernel_launch(void* const* d_in, const int* in_sizes, int n_in,
                              void* d_out, int out_size, void* d_ws, size_t ws_size,
                              hipStream_t stream) {
  (void)in_sizes; (void)n_in; (void)out_size; (void)ws_size;
  const float* x      = (const float*)d_in[0];
  const float* gate_w = (const float*)d_in[1];
  const float* w1     = (const float*)d_in[2];
  const float* w3     = (const float*)d_in[3];
  const float* w2     = (const float*)d_in[4];
  const float* sw1    = (const float*)d_in[5];
  const float* sw3    = (const float*)d_in[6];
  const float* sw2    = (const float*)d_in[7];
  float* out = (float*)d_out;

  char* p = (char*)d_ws;
  auto take = [&](size_t bytes) -> char* {
    char* q = p;
    p += (bytes + 255) & ~(size_t)255;
    return q;
  };
  __bf16* xb   = (__bf16*)take((size_t)T_TOKENS * DIM * 2);
  __bf16* w1b  = (__bf16*)take((size_t)N_EXPERTS * HID * DIM * 2);
  __bf16* w3b  = (__bf16*)take((size_t)N_EXPERTS * HID * DIM * 2);
  __bf16* w2b  = (__bf16*)take((size_t)N_EXPERTS * DIM * HID * 2);
  __bf16* sw1b = (__bf16*)take((size_t)HID * DIM * 2);
  __bf16* sw3b = (__bf16*)take((size_t)HID * DIM * 2);
  __bf16* sw2b = (__bf16*)take((size_t)DIM * HID * 2);
  int*   cnt    = (int*)take(N_EXPERTS * sizeof(int));
  int*   rowTok = (int*)take((size_t)N_EXPERTS * MAX_ROWS_PER_EXPERT * sizeof(int));
  float* rowW   = (float*)take((size_t)N_EXPERTS * MAX_ROWS_PER_EXPERT * sizeof(float));

  init_counts<<<1, 32, 0, stream>>>(cnt);

  cvt_f32_bf16<<<2048, 256, 0, stream>>>(x,   xb,   T_TOKENS * DIM);
  cvt_f32_bf16<<<4096, 256, 0, stream>>>(w1,  w1b,  N_EXPERTS * HID * DIM);
  cvt_f32_bf16<<<4096, 256, 0, stream>>>(w3,  w3b,  N_EXPERTS * HID * DIM);
  cvt_f32_bf16<<<4096, 256, 0, stream>>>(w2,  w2b,  N_EXPERTS * DIM * HID);
  cvt_f32_bf16<<<1024, 256, 0, stream>>>(sw1, sw1b, HID * DIM);
  cvt_f32_bf16<<<1024, 256, 0, stream>>>(sw3, sw3b, HID * DIM);
  cvt_f32_bf16<<<1024, 256, 0, stream>>>(sw2, sw2b, DIM * HID);

  router_kernel<<<T_TOKENS, 256, 0, stream>>>(x, gate_w, cnt, rowTok, rowW);

  // Shared path writes out (plain stores), then routed experts atomically add.
  shared_ffn_kernel<<<T_TOKENS / 16, 256, 0, stream>>>(xb, sw1b, sw3b, sw2b, out);
  moe_expert_kernel<<<dim3(T_TOKENS / 16, N_EXPERTS), 256, 0, stream>>>(
      xb, w1b, w3b, w2b, cnt, rowTok, rowW, out);
}
